// SparseAttention_23287312678977
// MI455X (gfx1250) — compile-verified, matrix operand order checked by reading, not run on hardware
//
#include <hip/hip_runtime.h>

#define T_DIM   2048
#define N_DIM   2
#define C_DIM   1024
#define H_DIM   16
#define D_HEAD  64
#define M_ROWS  (T_DIM * N_DIM)   // 4096 rows (t*N + n)
#define WIN     128
#define NEG_BIG (-1.0e9f)

typedef __attribute__((ext_vector_type(16))) _Float16 v16h;
typedef __attribute__((ext_vector_type(8)))  _Float16 v8h;
typedef __attribute__((ext_vector_type(8)))  float    v8f;

static __device__ __forceinline__ v8f wmma_f16(v16h a, v16h b, v8f c) {
  // D = A(16x32 f16) * B(32x16 f16) + C(16x16 f32)
  return __builtin_amdgcn_wmma_f32_16x16x32_f16(false, a, false, b, (short)0, c,
                                                false, false);
}

// A/B fragment slice: elems 0..7 = p[hi*8 ..], elems 8..15 = p[16 + hi*8 ..]
static __device__ __forceinline__ v16h load_frag_f16(const _Float16* p, int hi) {
  v8h a = *(const v8h*)(p + hi * 8);
  v8h b = *(const v8h*)(p + 16 + hi * 8);
  v16h r;
#pragma unroll
  for (int i = 0; i < 8; ++i) { r[i] = a[i]; r[8 + i] = b[i]; }
  return r;
}

// ---------------------------------------------------------------------------
// fp32 -> f16 bulk convert (one pass; HBM-bound).  n must be divisible by 2048.
// ---------------------------------------------------------------------------
__global__ __launch_bounds__(256) void cvt_kernel(const float* __restrict__ in,
                                                  _Float16* __restrict__ out) {
  const size_t i = ((size_t)blockIdx.x * 256 + threadIdx.x) * 8;
  v8f x = *(const v8f*)(in + i);
  v8h y;
#pragma unroll
  for (int j = 0; j < 8; ++j) y[j] = (_Float16)x[j];
  *(v8h*)(out + i) = y;
}

// ---------------------------------------------------------------------------
// f16 GEMM, 32x64 tile per wave: P = X * W^T + b  (rows m = t*N+n, cols c)
// MODE 0: store f16 head layout out[n][h][t][d]        (Q, K)
// MODE 1: store f16 transposed head layout out[n][h][d][t]   (V)
// grid = (C/64, M/32), 32 threads (1 wave) per block.
// ---------------------------------------------------------------------------
template <int MODE>
__global__ __launch_bounds__(32) void proj_kernel(
    const _Float16* __restrict__ X, const _Float16* __restrict__ W,
    const float* __restrict__ bias, _Float16* __restrict__ out) {
  const int lane = threadIdx.x & 31;
  const int lo = lane & 15, hi = lane >> 4;
  const int row0 = blockIdx.y * 32;
  const int col0 = blockIdx.x * 64;

  const _Float16* arow0 = X + (size_t)(row0 + lo) * C_DIM;
  const _Float16* arow1 = X + (size_t)(row0 + 16 + lo) * C_DIM;
  const _Float16* brow0 = W + (size_t)(col0 + lo) * C_DIM;
  const _Float16* brow1 = W + (size_t)(col0 + 16 + lo) * C_DIM;
  const _Float16* brow2 = W + (size_t)(col0 + 32 + lo) * C_DIM;
  const _Float16* brow3 = W + (size_t)(col0 + 48 + lo) * C_DIM;

  v8f acc[2][4] = {};
  for (int k0 = 0; k0 < C_DIM; k0 += 32) {
    if (k0 + 256 < C_DIM) {               // global_prefetch_b8 of next slices
      __builtin_prefetch(arow0 + k0 + 256, 0, 3);
      __builtin_prefetch(arow1 + k0 + 256, 0, 3);
      __builtin_prefetch(brow0 + k0 + 256, 0, 3);
      __builtin_prefetch(brow2 + k0 + 256, 0, 3);
    }
    const v16h a0 = load_frag_f16(arow0 + k0, hi);
    const v16h a1 = load_frag_f16(arow1 + k0, hi);
    const v16h b0 = load_frag_f16(brow0 + k0, hi);
    const v16h b1 = load_frag_f16(brow1 + k0, hi);
    const v16h b2 = load_frag_f16(brow2 + k0, hi);
    const v16h b3 = load_frag_f16(brow3 + k0, hi);
    acc[0][0] = wmma_f16(a0, b0, acc[0][0]);
    acc[1][0] = wmma_f16(a1, b0, acc[1][0]);
    acc[0][1] = wmma_f16(a0, b1, acc[0][1]);
    acc[1][1] = wmma_f16(a1, b1, acc[1][1]);
    acc[0][2] = wmma_f16(a0, b2, acc[0][2]);
    acc[1][2] = wmma_f16(a1, b2, acc[1][2]);
    acc[0][3] = wmma_f16(a0, b3, acc[0][3]);
    acc[1][3] = wmma_f16(a1, b3, acc[1][3]);
  }

#pragma unroll
  for (int g = 0; g < 4; ++g) {
    const int col = col0 + g * 16 + lo;   // output channel
    const float bv = bias[col];
    const int h = col >> 6;               // / D_HEAD
    const int d = col & 63;
#pragma unroll
    for (int mi = 0; mi < 2; ++mi) {
#pragma unroll
      for (int r = 0; r < 8; ++r) {
        const int m = row0 + mi * 16 + r + 8 * hi;  // row = t*N + n
        const int t = m >> 1;                       // N_DIM == 2
        const int n = m & 1;
        const float val = acc[mi][g][r] + bv;
        size_t idx;
        if (MODE == 0)
          idx = ((size_t)((n * H_DIM + h) * T_DIM + t)) * D_HEAD + d;
        else
          idx = ((size_t)((n * H_DIM + h) * D_HEAD + d)) * T_DIM + t;
        out[idx] = (_Float16)val;
      }
    }
  }
}

// ---------------------------------------------------------------------------
// Flash attention, one wave per (n, h, 16-query tile).
// S^T = K_tile * Q^T: the 16x16 f32 D-fragment of S^T (lane = query column,
// VGPR r = key row r+8*hi) is element-for-element the A-fragment layout of P
// for the P*V WMMA -> no cross-lane shuffle between the two GEMMs.
// Dead 32-key tiles (outside band, beyond key prefix) skipped scalar-uniformly.
// ---------------------------------------------------------------------------
__global__ __launch_bounds__(32) void attn_kernel(
    const _Float16* __restrict__ Qh, const _Float16* __restrict__ Kh,
    const _Float16* __restrict__ Vt, const int* __restrict__ key_length,
    _Float16* __restrict__ Y) {
  const int lane = threadIdx.x & 31;
  const int lo = lane & 15, hi = lane >> 4;
  const int qbase = blockIdx.x * 16;
  const int h = blockIdx.y;
  const int n = blockIdx.z;
  const int keylen = key_length[n];

  const _Float16* Qp = Qh + ((size_t)(n * H_DIM + h) * T_DIM) * D_HEAD;
  const _Float16* Kp = Kh + ((size_t)(n * H_DIM + h) * T_DIM) * D_HEAD;
  const _Float16* Vp = Vt + ((size_t)(n * H_DIM + h) * D_HEAD) * T_DIM;

  // Q^T B-fragments: lane column q = lo, two 32-wide d chunks.
  const _Float16* qrow = Qp + (size_t)(qbase + lo) * D_HEAD;
  const v16h qb0 = load_frag_f16(qrow, hi);        // d = 0..31
  const v16h qb1 = load_frag_f16(qrow + 32, hi);   // d = 32..63

  v8f o[4] = {};                 // O fragments: d-groups of 16
  float m_run = -INFINITY;
  float l_run = 0.0f;
  const int qg = qbase + lo;     // this lane's query row (dup across hi halves)

  for (int kb = 0; kb < T_DIM; kb += 32) {
    // tile fully masked iff no band overlap AND beyond the key prefix
    const bool band = (kb <= qbase + 15 + WIN) && (kb + 31 >= qbase - WIN);
    if (!band && kb >= keylen) continue;

    // K A-fragments (lane row = key), two row halves x two d chunks
    const _Float16* kr0 = Kp + (size_t)(kb + lo) * D_HEAD;
    const _Float16* kr1 = Kp + (size_t)(kb + 16 + lo) * D_HEAD;
    if (kb + 32 < T_DIM) {       // prefetch next key tile rows
      __builtin_prefetch(kr0 + 32 * D_HEAD, 0, 3);
      __builtin_prefetch(kr1 + 32 * D_HEAD, 0, 3);
    }
    v8f s0 = {}, s1 = {};
    s0 = wmma_f16(load_frag_f16(kr0, hi), qb0, s0);
    s0 = wmma_f16(load_frag_f16(kr0 + 32, hi), qb1, s0);
    s1 = wmma_f16(load_frag_f16(kr1, hi), qb0, s1);
    s1 = wmma_f16(load_frag_f16(kr1 + 32, hi), qb1, s1);

    // scale + mask + per-query max
    float mt = -INFINITY;
#pragma unroll
    for (int r = 0; r < 8; ++r) {
      const int kg0 = kb + r + 8 * hi;
      const int kg1 = kg0 + 16;
      float v0 = s0[r] * 0.125f;   // 1/sqrt(Dh)
      float v1 = s1[r] * 0.125f;
      int d0 = qg - kg0; d0 = d0 < 0 ? -d0 : d0;
      int d1 = qg - kg1; d1 = d1 < 0 ? -d1 : d1;
      if (!(d0 <= WIN || kg0 < keylen)) v0 = NEG_BIG;
      if (!(d1 <= WIN || kg1 < keylen)) v1 = NEG_BIG;
      s0[r] = v0; s1[r] = v1;
      mt = fmaxf(mt, fmaxf(v0, v1));
    }
    mt = fmaxf(mt, __shfl_xor(mt, 16, 32));   // join the two key halves
    const float m_new = fmaxf(m_run, mt);
    const float alpha = __expf(m_run - m_new);

    // exp -> P fragment (already in A-fragment layout), row-sum
    v16h pa;
    float rs = 0.0f;
#pragma unroll
    for (int r = 0; r < 8; ++r) {
      const float p0 = __expf(s0[r] - m_new);
      const float p1 = __expf(s1[r] - m_new);
      rs += p0 + p1;
      pa[r] = (_Float16)p0;
      pa[8 + r] = (_Float16)p1;
    }
    rs += __shfl_xor(rs, 16, 32);
    l_run = l_run * alpha + rs;
    m_run = m_new;

    // rescale O accumulators; O-fragment row q = r + 8*hi, alpha lives in lane q
#pragma unroll
    for (int r = 0; r < 8; ++r) {
      const float ar = __shfl(alpha, hi * 8 + r, 32);
      o[0][r] *= ar; o[1][r] *= ar; o[2][r] *= ar; o[3][r] *= ar;
    }

    // O += P * V  (V transposed in memory: Vp[d][t] contiguous in t)
#pragma unroll
    for (int g = 0; g < 4; ++g) {
      const _Float16* vrow = Vp + (size_t)(g * 16 + lo) * T_DIM + kb;
      o[g] = wmma_f16(pa, load_frag_f16(vrow, hi), o[g]);
    }
  }

  // epilogue: O/l  -> Y[(t*N+n)*C + h*64 + d]  (f16, feeds the out-projection)
#pragma unroll
  for (int r = 0; r < 8; ++r) {
    const float lr = __shfl(l_run, hi * 8 + r, 32);
    const float inv = 1.0f / lr;
    const int qrow_g = qbase + r + 8 * hi;
#pragma unroll
    for (int g = 0; g < 4; ++g) {
      const int c = h * D_HEAD + g * 16 + lo;
      Y[((size_t)qrow_g * N_DIM + n) * C_DIM + c] = (_Float16)(o[g][r] * inv);
    }
  }
}

// ---------------------------------------------------------------------------
// Output projection: out = Y(f16, MxC) * Wo^T(f16) + bo -> fp32 (T,N,C)
// Same 32x64 tile structure as proj_kernel.
// ---------------------------------------------------------------------------
__global__ __launch_bounds__(32) void outproj_kernel(
    const _Float16* __restrict__ Yin, const _Float16* __restrict__ W,
    const float* __restrict__ bias, float* __restrict__ out) {
  const int lane = threadIdx.x & 31;
  const int lo = lane & 15, hi = lane >> 4;
  const int row0 = blockIdx.y * 32;
  const int col0 = blockIdx.x * 64;

  const _Float16* arow0 = Yin + (size_t)(row0 + lo) * C_DIM;
  const _Float16* arow1 = Yin + (size_t)(row0 + 16 + lo) * C_DIM;
  const _Float16* brow0 = W + (size_t)(col0 + lo) * C_DIM;
  const _Float16* brow1 = W + (size_t)(col0 + 16 + lo) * C_DIM;
  const _Float16* brow2 = W + (size_t)(col0 + 32 + lo) * C_DIM;
  const _Float16* brow3 = W + (size_t)(col0 + 48 + lo) * C_DIM;

  v8f acc[2][4] = {};
  for (int k0 = 0; k0 < C_DIM; k0 += 32) {
    if (k0 + 256 < C_DIM) {
      __builtin_prefetch(arow0 + k0 + 256, 0, 3);
      __builtin_prefetch(arow1 + k0 + 256, 0, 3);
      __builtin_prefetch(brow0 + k0 + 256, 0, 3);
      __builtin_prefetch(brow2 + k0 + 256, 0, 3);
    }
    const v16h a0 = load_frag_f16(arow0 + k0, hi);
    const v16h a1 = load_frag_f16(arow1 + k0, hi);
    const v16h b0 = load_frag_f16(brow0 + k0, hi);
    const v16h b1 = load_frag_f16(brow1 + k0, hi);
    const v16h b2 = load_frag_f16(brow2 + k0, hi);
    const v16h b3 = load_frag_f16(brow3 + k0, hi);
    acc[0][0] = wmma_f16(a0, b0, acc[0][0]);
    acc[1][0] = wmma_f16(a1, b0, acc[1][0]);
    acc[0][1] = wmma_f16(a0, b1, acc[0][1]);
    acc[1][1] = wmma_f16(a1, b1, acc[1][1]);
    acc[0][2] = wmma_f16(a0, b2, acc[0][2]);
    acc[1][2] = wmma_f16(a1, b2, acc[1][2]);
    acc[0][3] = wmma_f16(a0, b3, acc[0][3]);
    acc[1][3] = wmma_f16(a1, b3, acc[1][3]);
  }

#pragma unroll
  for (int g = 0; g < 4; ++g) {
    const int col = col0 + g * 16 + lo;
    const float bv = bias[col];
#pragma unroll
    for (int mi = 0; mi < 2; ++mi) {
#pragma unroll
      for (int r = 0; r < 8; ++r) {
        const int m = row0 + mi * 16 + r + 8 * hi;
        out[(size_t)m * C_DIM + col] = acc[mi][g][r] + bv;
      }
    }
  }
}

// ---------------------------------------------------------------------------
extern "C" void kernel_launch(void* const* d_in, const int* in_sizes, int n_in,
                              void* d_out, int out_size, void* d_ws,
                              size_t ws_size, hipStream_t stream) {
  (void)in_sizes; (void)n_in; (void)out_size; (void)ws_size;
  const float* q  = (const float*)d_in[0];
  const float* k  = (const float*)d_in[1];
  const float* v  = (const float*)d_in[2];
  const float* Wq = (const float*)d_in[3];
  const float* bq = (const float*)d_in[4];
  const float* Wk = (const float*)d_in[5];
  const float* bk = (const float*)d_in[6];
  const float* Wv = (const float*)d_in[7];
  const float* bv = (const float*)d_in[8];
  const float* Wo = (const float*)d_in[9];
  const float* bo = (const float*)d_in[10];
  const int* key_length = (const int*)d_in[11];
  float* out = (float*)d_out;

  // Workspace (f16 elements), total 64 MB:
  //   Qh, Kh, Vt, Y          : 4 x N*H*T*Dh  = 4 x 4M   (32 MB)
  //   q16, k16, v16          : 3 x T*N*C     = 3 x 4M   (24 MB)
  //   Wq16, Wk16, Wv16, Wo16 : 4 x C*C       = 4 x 1M   ( 8 MB)
  const size_t HSZ = (size_t)N_DIM * H_DIM * T_DIM * D_HEAD;  // 4,194,304
  const size_t WSZ = (size_t)C_DIM * C_DIM;                   // 1,048,576
  _Float16* Qh   = (_Float16*)d_ws;
  _Float16* Kh   = Qh + HSZ;
  _Float16* Vt   = Kh + HSZ;
  _Float16* Y    = Vt + HSZ;
  _Float16* q16  = Y + HSZ;
  _Float16* k16  = q16 + HSZ;
  _Float16* v16  = k16 + HSZ;
  _Float16* Wq16 = v16 + HSZ;
  _Float16* Wk16 = Wq16 + WSZ;
  _Float16* Wv16 = Wk16 + WSZ;
  _Float16* Wo16 = Wv16 + WSZ;

  // 1) bulk fp32 -> f16 conversions (removes all cvt work from GEMM hot loops)
  cvt_kernel<<<HSZ / 2048, 256, 0, stream>>>(q, q16);
  cvt_kernel<<<HSZ / 2048, 256, 0, stream>>>(k, k16);
  cvt_kernel<<<HSZ / 2048, 256, 0, stream>>>(v, v16);
  cvt_kernel<<<WSZ / 2048, 256, 0, stream>>>(Wq, Wq16);
  cvt_kernel<<<WSZ / 2048, 256, 0, stream>>>(Wk, Wk16);
  cvt_kernel<<<WSZ / 2048, 256, 0, stream>>>(Wv, Wv16);
  cvt_kernel<<<WSZ / 2048, 256, 0, stream>>>(Wo, Wo16);

  // 2) input projections into WMMA-friendly head layouts
  const dim3 gp(C_DIM / 64, M_ROWS / 32);  // (16, 128) 32x64 tiles, 1 wave each
  proj_kernel<0><<<gp, 32, 0, stream>>>(q16, Wq16, bq, Qh);
  proj_kernel<0><<<gp, 32, 0, stream>>>(k16, Wk16, bk, Kh);
  proj_kernel<1><<<gp, 32, 0, stream>>>(v16, Wv16, bv, Vt);

  // 3) banded + prefix-masked flash attention
  attn_kernel<<<dim3(T_DIM / 16, H_DIM, N_DIM), 32, 0, stream>>>(
      Qh, Kh, Vt, key_length, Y);

  // 4) output projection (fp32 result + bias)
  outproj_kernel<<<gp, 32, 0, stream>>>(Y, Wo16, bo, out);
}
